// TopKSAE_27496380629359
// MI455X (gfx1250) — matrix operand order repaired and checked by reading, not compiled, not run on hardware
//
#include <hip/hip_runtime.h>

// ---------------------------------------------------------------------------
// TopK SAE forward for MI455X (gfx1250), wave32 + v_wmma_f32_16x16x32_bf16.
//   x_recon == x (straight-through), z = x @ w_enc^T + b, z_bar = top64 mask.
// GEMM: block = 256 rows x 48 cols, 8 waves; wave tile 32x48 (2A x 3B frags,
// 6 f32 accumulators) -> 26.7 B of L2-resident loads per WMMA.
// ---------------------------------------------------------------------------

#define N_DIM 4096
#define D_DIM 768
#define H_DIM 30000
#define K_TOP 64

typedef __bf16        v16bf __attribute__((ext_vector_type(16)));
typedef float         v8f   __attribute__((ext_vector_type(8)));
typedef unsigned int  v8u   __attribute__((ext_vector_type(8)));

__device__ __forceinline__ unsigned short f2bf(float f) {
    unsigned u = __builtin_bit_cast(unsigned int, f);
    unsigned r = u + 0x7FFFu + ((u >> 16) & 1u);   // round-to-nearest-even
    return (unsigned short)(r >> 16);
}

// monotonic key: ordering of keys == ordering of floats
__device__ __forceinline__ unsigned fkey(float f) {
    unsigned u = __builtin_bit_cast(unsigned int, f);
    return (u & 0x80000000u) ? ~u : (u | 0x80000000u);
}

// 16 bf16 from two 16B chunks at p and p+16 elements (A-fragment pattern)
__device__ __forceinline__ v16bf load_fragA(const unsigned short* p) {
    uint4 lo = *(const uint4*)(p);
    uint4 hi = *(const uint4*)(p + 16);
    v8u u; u[0]=lo.x; u[1]=lo.y; u[2]=lo.z; u[3]=lo.w;
           u[4]=hi.x; u[5]=hi.y; u[6]=hi.z; u[7]=hi.w;
    return __builtin_bit_cast(v16bf, u);
}

// 16 contiguous bf16 (B-fragment pattern: one 32B run)
__device__ __forceinline__ v16bf load_fragB(const unsigned short* p) {
    uint4 lo = *(const uint4*)(p);
    uint4 hi = *(const uint4*)(p + 8);
    v8u u; u[0]=lo.x; u[1]=lo.y; u[2]=lo.z; u[3]=lo.w;
           u[4]=hi.x; u[5]=hi.y; u[6]=hi.z; u[7]=hi.w;
    return __builtin_bit_cast(v16bf, u);
}

// ---------------------------------------------------------------------------
// K0: zero the per-row histograms
__global__ void sae_zero_i32(int* __restrict__ p, int n) {
    int i = blockIdx.x * blockDim.x + threadIdx.x;
    if (i < n) p[i] = 0;
}

// K1: x -> bf16 (for WMMA A) and x_recon = x (straight-through forward value)
__global__ void sae_prep_x(const float* __restrict__ x,
                           unsigned short* __restrict__ xbf,
                           float* __restrict__ xrecon, int n) {
    int i = blockIdx.x * blockDim.x + threadIdx.x;
    if (i < n) { float v = x[i]; xbf[i] = f2bf(v); xrecon[i] = v; }
}

// K2: weight-norm encoder rows -> bf16   w[h,:] = v[h,:] * g[h]/||v[h,:]||
__global__ __launch_bounds__(256)
void sae_norm_encoder(const float* __restrict__ v_enc,
                      const float* __restrict__ g_enc,
                      unsigned short* __restrict__ wbf) {
    const int h   = blockIdx.x;
    const int tid = threadIdx.x;
    const float* row = v_enc + (size_t)h * D_DIM;
    float vals[3]; float s = 0.f;
    #pragma unroll
    for (int i = 0; i < 3; ++i) { vals[i] = row[tid + i * 256]; s += vals[i] * vals[i]; }
    __shared__ float red[256];
    red[tid] = s; __syncthreads();
    for (int st = 128; st > 0; st >>= 1) {
        if (tid < st) red[tid] += red[tid + st];
        __syncthreads();
    }
    const float scale = g_enc[h] * rsqrtf(red[0]);
    unsigned short* out = wbf + (size_t)h * D_DIM;
    #pragma unroll
    for (int i = 0; i < 3; ++i) out[tid + i * 256] = f2bf(vals[i] * scale);
}

// ---------------------------------------------------------------------------
// K3: z = x @ w_enc^T + b  via v_wmma_f32_16x16x32_bf16
//     Block tile 256(M) x 48(H): 8 waves, each 32 rows x 48 cols
//     (2 A-frags x 3 B-frags -> 6 independent accumulator chains).
//     Epilogue fuses bias add, z store, and top-byte key histogram atomics.
//     Working set (wbf 46MB + xbf 6MB) is L2-resident on the 192MB L2.
__global__ __launch_bounds__(256)
void sae_encoder_gemm(const unsigned short* __restrict__ xbf,
                      const unsigned short* __restrict__ wbf,
                      const float* __restrict__ bias,
                      float* __restrict__ z,
                      int* __restrict__ hist) {
    const int wave = threadIdx.x >> 5;     // 0..7 (wave32)
    const int lane = threadIdx.x & 31;
    const int hsel = lane >> 4;            // half-wave select
    const int m16  = lane & 15;

    const int rowA0 = blockIdx.x * 256 + wave * 32 + m16;   // M group 0
    const int colB  = blockIdx.y * 48;

    const unsigned short* aRow0 = xbf + (size_t)rowA0 * D_DIM;
    const unsigned short* aRow1 = aRow0 + (size_t)16 * D_DIM;
    const unsigned short* b0    = wbf + (size_t)(colB +  0 + m16) * D_DIM;
    const unsigned short* b1    = wbf + (size_t)(colB + 16 + m16) * D_DIM;
    const unsigned short* b2    = wbf + (size_t)(colB + 32 + m16) * D_DIM;

    v8f acc[2][3];
    #pragma unroll
    for (int g = 0; g < 2; ++g)
        #pragma unroll
        for (int j = 0; j < 3; ++j) acc[g][j] = (v8f){};

    #pragma unroll 4
    for (int k0 = 0; k0 < D_DIM; k0 += 32) {
        // A 16x32: lanes 0-15 hold K 0..15,
        //          lanes 16-31 hold K 16..31 (matches B)
        v16bf aF0 = load_fragB(aRow0 + k0 + hsel * 16);
        v16bf aF1 = load_fragB(aRow1 + k0 + hsel * 16);
        // B 32x16: lane m16 holds column, half-waves split K 0..15 / 16..31
        v16bf bF0 = load_fragB(b0 + k0 + hsel * 16);
        v16bf bF1 = load_fragB(b1 + k0 + hsel * 16);
        v16bf bF2 = load_fragB(b2 + k0 + hsel * 16);
        if (k0 + 32 < D_DIM) {   // keep next K-slab warm (global_prefetch_b8)
            __builtin_prefetch(aRow0 + k0 + 32, 0, 3);
            __builtin_prefetch(aRow1 + k0 + 32, 0, 3);
            __builtin_prefetch(b0 + k0 + 32, 0, 3);
            __builtin_prefetch(b1 + k0 + 32, 0, 3);
            __builtin_prefetch(b2 + k0 + 32, 0, 3);
        }
        acc[0][0] = __builtin_amdgcn_wmma_f32_16x16x32_bf16(false, aF0, false, bF0,
                                                            (short)0, acc[0][0], false, false);
        acc[0][1] = __builtin_amdgcn_wmma_f32_16x16x32_bf16(false, aF0, false, bF1,
                                                            (short)0, acc[0][1], false, false);
        acc[0][2] = __builtin_amdgcn_wmma_f32_16x16x32_bf16(false, aF0, false, bF2,
                                                            (short)0, acc[0][2], false, false);
        acc[1][0] = __builtin_amdgcn_wmma_f32_16x16x32_bf16(false, aF1, false, bF0,
                                                            (short)0, acc[1][0], false, false);
        acc[1][1] = __builtin_amdgcn_wmma_f32_16x16x32_bf16(false, aF1, false, bF1,
                                                            (short)0, acc[1][1], false, false);
        acc[1][2] = __builtin_amdgcn_wmma_f32_16x16x32_bf16(false, aF1, false, bF2,
                                                            (short)0, acc[1][2], false, false);
    }

    // C layout: VGPR i holds row (i) for lanes 0-15, row (i+8) for lanes 16-31,
    // column = lane & 15.
    const int rowC = blockIdx.x * 256 + wave * 32;
    #pragma unroll
    for (int g = 0; g < 2; ++g) {
        #pragma unroll
        for (int j = 0; j < 3; ++j) {
            const v8f a = acc[g][j];
            const int col = colB + j * 16 + m16;
            const float bz = bias[col];
            #pragma unroll
            for (int i = 0; i < 8; ++i) {
                const int r = rowC + g * 16 + i + hsel * 8;
                const float v = a[i] + bz;
                z[(size_t)r * H_DIM + col] = v;
                atomicAdd(&hist[r * 256 + (int)(fkey(v) >> 24)], 1);
            }
        }
    }
}

// ---------------------------------------------------------------------------
// K4: per-row exact 64th-largest threshold. One block per row.
//     Find histogram bin holding the K-th value, gather all candidates
//     >= bin floor into LDS, bitonic-sort descending, take element K-1.
__global__ __launch_bounds__(256)
void sae_topk_threshold(const float* __restrict__ z,
                        const int* __restrict__ hist,
                        float* __restrict__ thr) {
    const int row = blockIdx.x;
    const int tid = threadIdx.x;
    __shared__ int      sh_hist[256];
    __shared__ float    cand[1024];
    __shared__ int      scnt;
    __shared__ unsigned s_binlow;

    sh_hist[tid] = hist[row * 256 + tid];
    if (tid == 0) scnt = 0;
    __syncthreads();

    if (tid == 0) {
        int acc = 0, b = 0;
        for (int i = 255; i >= 0; --i) {
            acc += sh_hist[i];
            if (acc >= K_TOP) { b = i; break; }
        }
        s_binlow = ((unsigned)b) << 24;
    }
    __syncthreads();
    const unsigned binlow = s_binlow;

    const float* zr = z + (size_t)row * H_DIM;
    for (int i = tid; i < H_DIM; i += 256) {
        float v = zr[i];
        if (fkey(v) >= binlow) {
            int p = atomicAdd(&scnt, 1);
            if (p < 1024) cand[p] = v;
        }
    }
    __syncthreads();
    const int n = scnt < 1024 ? scnt : 1024;
    for (int i = tid; i < 1024; i += 256)
        if (i >= n) cand[i] = -3.402823466e38f;
    __syncthreads();

    // bitonic sort, overall descending
    for (int k = 2; k <= 1024; k <<= 1) {
        for (int j = k >> 1; j > 0; j >>= 1) {
            for (int i = tid; i < 1024; i += 256) {
                int ixj = i ^ j;
                if (ixj > i) {
                    bool asc = (i & k) != 0;          // (i&k)==0 segments: descending
                    float a = cand[i], b = cand[ixj];
                    if (asc ? (a > b) : (a < b)) { cand[i] = b; cand[ixj] = a; }
                }
            }
            __syncthreads();
        }
    }
    if (tid == 0) thr[row] = cand[K_TOP - 1];
}

// K5: z_bar = z * (z >= thr_row), vectorized float4 (H % 4 == 0)
__global__ __launch_bounds__(256)
void sae_apply_topk(const float* __restrict__ z,
                    const float* __restrict__ thr,
                    float* __restrict__ zbar) {
    const size_t i = (size_t)blockIdx.x * blockDim.x + threadIdx.x;  // float4 idx
    const float4 v = ((const float4*)z)[i];
    const int row = (int)(i / (H_DIM / 4));
    const float t = thr[row];
    float4 o;
    o.x = (v.x >= t) ? v.x : 0.f;
    o.y = (v.y >= t) ? v.y : 0.f;
    o.z = (v.z >= t) ? v.z : 0.f;
    o.w = (v.w >= t) ? v.w : 0.f;
    ((float4*)zbar)[i] = o;
}

// ---------------------------------------------------------------------------
extern "C" void kernel_launch(void* const* d_in, const int* in_sizes, int n_in,
                              void* d_out, int out_size, void* d_ws, size_t ws_size,
                              hipStream_t stream) {
    (void)in_sizes; (void)n_in; (void)out_size; (void)ws_size;
    const float* x     = (const float*)d_in[0];
    const float* v_enc = (const float*)d_in[1];
    const float* g_enc = (const float*)d_in[2];
    const float* b_enc = (const float*)d_in[3];
    // d_in[4] (v_dec) is unused: straight-through makes x_recon's forward
    // value identically x, so the decoder GEMM contributes nothing here.

    float* out   = (float*)d_out;
    float* xrec  = out;                                   // [N, D]
    float* z     = out + (size_t)N_DIM * D_DIM;           // [N, H]
    float* zbar  = z   + (size_t)N_DIM * H_DIM;           // [N, H]

    char* ws = (char*)d_ws;
    size_t off = 0;
    unsigned short* xbf = (unsigned short*)(ws + off);
    off += ((size_t)N_DIM * D_DIM * 2 + 255) & ~(size_t)255;
    unsigned short* wbf = (unsigned short*)(ws + off);
    off += ((size_t)H_DIM * D_DIM * 2 + 255) & ~(size_t)255;
    int* hist = (int*)(ws + off);
    off += ((size_t)N_DIM * 256 * 4 + 255) & ~(size_t)255;
    float* thr = (float*)(ws + off);

    const int nHist = N_DIM * 256;
    sae_zero_i32<<<(nHist + 255) / 256, 256, 0, stream>>>(hist, nHist);

    const int nX = N_DIM * D_DIM;
    sae_prep_x<<<(nX + 255) / 256, 256, 0, stream>>>(x, xbf, xrec, nX);

    sae_norm_encoder<<<H_DIM, 256, 0, stream>>>(v_enc, g_enc, wbf);

    dim3 gemmGrid(N_DIM / 256, H_DIM / 48);               // 16 x 625
    sae_encoder_gemm<<<gemmGrid, 256, 0, stream>>>(xbf, wbf, b_enc, z, hist);

    sae_topk_threshold<<<N_DIM, 256, 0, stream>>>(z, hist, thr);

    const size_t nQuads = (size_t)N_DIM * H_DIM / 4;
    sae_apply_topk<<<(unsigned)(nQuads / 256), 256, 0, stream>>>(z, thr, zbar);
}
